// MultiHeadAttention_15796889715425
// MI455X (gfx1250) — matrix operand order repaired and checked by reading, not compiled, not run on hardware
//
#include <hip/hip_runtime.h>
#include <hip/hip_bf16.h>
#include <math.h>

// ---------------------------------------------------------------- types ----
typedef __attribute__((ext_vector_type(16))) __bf16 bf16x16;
typedef __attribute__((ext_vector_type(8)))  __bf16 bf16x8;
typedef __attribute__((ext_vector_type(8)))  float  f32x8;

#define D_MODEL 1024
#define SEQ     2048
#define BATCH   4
#define NHEAD   16
#define DK      64

// ------------------------------------------------------------ fp32->bf16 --
__global__ void cvt_f32_bf16(const float* __restrict__ src,
                             __bf16* __restrict__ dst, int n) {
    int i = blockIdx.x * blockDim.x + threadIdx.x;
    if (i < n) dst[i] = (__bf16)src[i];
}

// ------------------------------------------------- DPP 16-lane reductions --
// row_ror:n rotates within each 16-lane row (wave32): circular reduction
// over the 16-lane half without touching the LDS unit.
template <int CTRL>
__device__ __forceinline__ float dpp_mov_f32(float x) {
    return __int_as_float(
        __builtin_amdgcn_update_dpp(0, __float_as_int(x), CTRL, 0xF, 0xF, true));
}
__device__ __forceinline__ float halfmax16(float x) {
    x = fmaxf(x, dpp_mov_f32<0x121>(x));   // row_ror:1
    x = fmaxf(x, dpp_mov_f32<0x122>(x));   // row_ror:2
    x = fmaxf(x, dpp_mov_f32<0x124>(x));   // row_ror:4
    x = fmaxf(x, dpp_mov_f32<0x128>(x));   // row_ror:8
    return x;
}
__device__ __forceinline__ float halfsum16(float x) {
    x += dpp_mov_f32<0x121>(x);
    x += dpp_mov_f32<0x122>(x);
    x += dpp_mov_f32<0x124>(x);
    x += dpp_mov_f32<0x128>(x);
    return x;
}

// -------------------------------------------------------------- GEMM ------
// C[m,n] = (sum_k A[m,k] * W[n,k] + bias[n]) * scale
// A: (M,K) bf16 row-major; W: (N,K) bf16 row-major (torch Linear weight).
// Block: 256 threads = 8 waves (2 along M, 4 along N). Wave tile: 32M x 64N.
// Grid: (M/64, N/256).
template <bool BF16OUT>
__global__ void gemm_xwT(const __bf16* __restrict__ A,
                         const __bf16* __restrict__ W,
                         const float*  __restrict__ bias,
                         void* __restrict__ Cout,
                         int M, int N, int K, float scale) {
    const int lane = threadIdx.x & 31;
    const int w    = threadIdx.x >> 5;            // 0..7
    const int lm   = lane & 15;
    const int half = lane >> 4;
    const int mw   = blockIdx.x * 64  + (w & 1) * 32;
    const int nw   = blockIdx.y * 256 + (w >> 1) * 64;

    const __bf16* Arow0 = A + (size_t)(mw + lm) * K;
    const __bf16* Arow1 = Arow0 + (size_t)16 * K;
    f32x8 acc[2][4] = {};

    for (int k0 = 0; k0 < K; k0 += 32) {
        // two A fragments: rows mw..mw+15 and mw+16..mw+31
        bf16x16 a0, a1;
        {
            bf16x8 lo = *(const bf16x8*)(Arow0 + k0 + half * 8);
            bf16x8 hi = *(const bf16x8*)(Arow0 + k0 + 16 + half * 8);
#pragma unroll
            for (int i = 0; i < 8; i++) { a0[i] = lo[i]; a0[8 + i] = hi[i]; }
        }
        {
            bf16x8 lo = *(const bf16x8*)(Arow1 + k0 + half * 8);
            bf16x8 hi = *(const bf16x8*)(Arow1 + k0 + 16 + half * 8);
#pragma unroll
            for (int i = 0; i < 8; i++) { a1[i] = lo[i]; a1[8 + i] = hi[i]; }
        }
#pragma unroll
        for (int t = 0; t < 4; t++) {
            const __bf16* wp = W + (size_t)(nw + t * 16 + lm) * K + k0 + half * 8;
            bf16x8 blo = *(const bf16x8*)(wp);
            bf16x8 bhi = *(const bf16x8*)(wp + 16);
            bf16x16 b;
#pragma unroll
            for (int i = 0; i < 8; i++) { b[i] = blo[i]; b[8 + i] = bhi[i]; }
            acc[0][t] = __builtin_amdgcn_wmma_f32_16x16x32_bf16(
                false, a0, false, b, (short)0, acc[0][t], false, false);
            acc[1][t] = __builtin_amdgcn_wmma_f32_16x16x32_bf16(
                false, a1, false, b, (short)0, acc[1][t], false, false);
        }
    }

#pragma unroll
    for (int r = 0; r < 2; r++) {
#pragma unroll
        for (int t = 0; t < 4; t++) {
            int n = nw + t * 16 + lm;
            float bv = bias[n];
#pragma unroll
            for (int v = 0; v < 8; v++) {
                int m = mw + r * 16 + v + 8 * half;
                float rr = (acc[r][t][v] + bv) * scale;
                if constexpr (BF16OUT)
                    ((__bf16*)Cout)[(size_t)m * N + n] = (__bf16)rr;
                else
                    ((float*)Cout)[(size_t)m * N + n] = rr;
            }
        }
    }
}

// --------------------------------------------------------- attention ------
// Flash-style online-softmax attention. One workgroup = 4 waves handles
// (b, h, 64 queries); each wave owns 16 queries. Scale is pre-folded into Q.
__global__ void attn_kernel(const __bf16* __restrict__ Q,
                            const __bf16* __restrict__ Kk,
                            const __bf16* __restrict__ V,
                            __bf16* __restrict__ ctx) {
    const int b  = blockIdx.z;
    const int hh = blockIdx.y;
    const int q0 = blockIdx.x * 64;
    const int w    = threadIdx.x >> 5;
    const int lane = threadIdx.x & 31;
    const int lm   = lane & 15;
    const int half = lane >> 4;
    const int qw   = q0 + w * 16;
    const int tid  = threadIdx.x;

    // padded rows to avoid LDS bank conflicts on b128 accesses
    __shared__ __bf16 Kt[32][72];      // [key][d]
    __shared__ __bf16 Vt[64][40];      // [d][key]   (transposed)
    __shared__ __bf16 Pw[4][16][40];   // per-wave P scratch [q][key]

    const size_t headoff = ((size_t)b * SEQ) * D_MODEL + hh * DK;
    const __bf16* Qh = Q  + headoff;
    const __bf16* Kh = Kk + headoff;
    const __bf16* Vh = V  + headoff;

    // Q fragments: 16x64 = two 16x32 A fragments
    bf16x16 qf[2];
    {
        const __bf16* qrow = Qh + (size_t)(qw + lm) * D_MODEL;
#pragma unroll
        for (int f = 0; f < 2; f++) {
            bf16x8 lo = *(const bf16x8*)(qrow + f * 32 + half * 8);
            bf16x8 hi = *(const bf16x8*)(qrow + f * 32 + 16 + half * 8);
#pragma unroll
            for (int i = 0; i < 8; i++) { qf[f][i] = lo[i]; qf[f][8 + i] = hi[i]; }
        }
    }

    float mrow[8], lsum[8];
    f32x8 acc[4] = {};
#pragma unroll
    for (int v = 0; v < 8; v++) { mrow[v] = -INFINITY; lsum[v] = 0.0f; }

    const int key = tid >> 2, ch = tid & 3;   // cooperative staging coords

    for (int kt = 0; kt < SEQ; kt += 32) {
        __syncthreads();   // previous-iteration Kt/Vt readers done
        {
            // K tile: async DMA global -> LDS (ASYNCcnt), 2x16B per thread
            const __bf16* ks = Kh + (size_t)(kt + key) * D_MODEL + ch * 16;
            uint32_t la = (uint32_t)(uintptr_t)&Kt[key][ch * 16];
            uint64_t ga = (uint64_t)(uintptr_t)ks;
            asm volatile("global_load_async_to_lds_b128 %0, %1, off"
                         :: "v"(la), "v"(ga) : "memory");
            uint32_t la2 = la + 16;
            uint64_t ga2 = ga + 16;
            asm volatile("global_load_async_to_lds_b128 %0, %1, off"
                         :: "v"(la2), "v"(ga2) : "memory");

            // V tile: load + transpose into LDS through VGPRs
            const __bf16* vs = Vh + (size_t)(kt + key) * D_MODEL + ch * 16;
            bf16x8 v0 = *(const bf16x8*)(vs);
            bf16x8 v1 = *(const bf16x8*)(vs + 8);
#pragma unroll
            for (int i = 0; i < 8; i++) {
                Vt[ch * 16 + i][key]     = v0[i];
                Vt[ch * 16 + 8 + i][key] = v1[i];
            }
            if (kt + 32 < SEQ) {   // L2 prefetch of next tile
                __builtin_prefetch(Kh + (size_t)(kt + 32 + key) * D_MODEL + ch * 16, 0, 1);
                __builtin_prefetch(Vh + (size_t)(kt + 32 + key) * D_MODEL + ch * 16, 0, 1);
            }
        }
        asm volatile("s_wait_asynccnt 0x0" ::: "memory");
        __syncthreads();

        // scores: S = Q(16x64) @ K^T -> two 16x16 C tiles (keys 0-15, 16-31)
        f32x8 c0 = {}, c1 = {};
#pragma unroll
        for (int f = 0; f < 2; f++) {
            const __bf16* kp0 = &Kt[lm][f * 32 + half * 8];
            bf16x8 l0 = *(const bf16x8*)(kp0);
            bf16x8 h0 = *(const bf16x8*)(kp0 + 16);
            bf16x16 b0;
#pragma unroll
            for (int i = 0; i < 8; i++) { b0[i] = l0[i]; b0[8 + i] = h0[i]; }
            c0 = __builtin_amdgcn_wmma_f32_16x16x32_bf16(
                false, qf[f], false, b0, (short)0, c0, false, false);

            const __bf16* kp1 = &Kt[16 + lm][f * 32 + half * 8];
            bf16x8 l1 = *(const bf16x8*)(kp1);
            bf16x8 h1 = *(const bf16x8*)(kp1 + 16);
            bf16x16 b1;
#pragma unroll
            for (int i = 0; i < 8; i++) { b1[i] = l1[i]; b1[8 + i] = h1[i]; }
            c1 = __builtin_amdgcn_wmma_f32_16x16x32_bf16(
                false, qf[f], false, b1, (short)0, c1, false, false);
        }

        // online softmax in C layout: lane owns row (v + 8*half), col lm
#pragma unroll
        for (int v = 0; v < 8; v++) {
            float r = halfmax16(fmaxf(c0[v], c1[v]));
            float mnew = fmaxf(mrow[v], r);
            float p0 = __expf(c0[v] - mnew);
            float p1 = __expf(c1[v] - mnew);
            float s  = halfsum16(p0 + p1);
            float alpha = __expf(mrow[v] - mnew);
            lsum[v] = lsum[v] * alpha + s;
            mrow[v] = mnew;
#pragma unroll
            for (int c = 0; c < 4; c++) acc[c][v] *= alpha;
            int row = v + 8 * half;
            Pw[w][row][lm]      = (__bf16)p0;
            Pw[w][row][16 + lm] = (__bf16)p1;
        }
        // NOTE: no barrier needed here. Pw is strictly per-wave scratch and
        // same-wave LDS ops complete in order (DScnt); the compiler inserts
        // the s_wait_dscnt before the dependent ds_loads below.

        // P as A fragment (16q x 32k)
        bf16x16 pf;
        {
            const __bf16* pr = &Pw[w][lm][0];
            bf16x8 lo = *(const bf16x8*)(pr + half * 8);
            bf16x8 hi = *(const bf16x8*)(pr + 16 + half * 8);
#pragma unroll
            for (int i = 0; i < 8; i++) { pf[i] = lo[i]; pf[8 + i] = hi[i]; }
        }

        // context += P @ V : four 16x16 d-chunk accumulators
#pragma unroll
        for (int c = 0; c < 4; c++) {
            const __bf16* vr = &Vt[c * 16 + lm][half * 8];
            bf16x8 lo = *(const bf16x8*)(vr);
            bf16x8 hi = *(const bf16x8*)(vr + 16);
            bf16x16 bv;
#pragma unroll
            for (int i = 0; i < 8; i++) { bv[i] = lo[i]; bv[8 + i] = hi[i]; }
            acc[c] = __builtin_amdgcn_wmma_f32_16x16x32_bf16(
                false, pf, false, bv, (short)0, acc[c], false, false);
        }
    }

    // normalize + write context in (B,S,D_MODEL) bf16 layout
#pragma unroll
    for (int v = 0; v < 8; v++) {
        float inv = 1.0f / lsum[v];
        int q = qw + v + 8 * half;
        __bf16* dst = ctx + ((size_t)b * SEQ + q) * D_MODEL + hh * DK;
#pragma unroll
        for (int c = 0; c < 4; c++)
            dst[c * 16 + lm] = (__bf16)(acc[c][v] * inv);
    }
}

// -------------------------------------------------------------- launch ----
extern "C" void kernel_launch(void* const* d_in, const int* in_sizes, int n_in,
                              void* d_out, int out_size, void* d_ws, size_t ws_size,
                              hipStream_t stream) {
    const float* query = (const float*)d_in[0];
    const float* key_  = (const float*)d_in[1];
    const float* value = (const float*)d_in[2];
    const float* W_Q   = (const float*)d_in[3];
    const float* b_Q   = (const float*)d_in[4];
    const float* W_K   = (const float*)d_in[5];
    const float* b_K   = (const float*)d_in[6];
    const float* W_V   = (const float*)d_in[7];
    const float* b_V   = (const float*)d_in[8];
    const float* W_O   = (const float*)d_in[9];
    const float* b_O   = (const float*)d_in[10];

    const int M  = BATCH * SEQ;            // 8192
    const int NB = M * D_MODEL;            // 8,388,608 activation elements
    const int NW = D_MODEL * D_MODEL;      // 1,048,576 weight elements

    char* ws = (char*)d_ws;
    const size_t BIG = (size_t)NB * sizeof(__bf16);   // 16 MiB
    __bf16* B0 = (__bf16*)(ws);
    __bf16* B1 = (__bf16*)(ws + BIG);
    __bf16* B2 = (__bf16*)(ws + 2 * BIG);
    __bf16* B3 = (__bf16*)(ws + 3 * BIG);
    __bf16* WQ = (__bf16*)(ws + 4 * BIG);
    __bf16* WK = WQ + NW;
    __bf16* WV = WK + NW;
    __bf16* WO = WV + NW;

    const int CT = 256;
    cvt_f32_bf16<<<(NB + CT - 1) / CT, CT, 0, stream>>>(query, B0, NB);
    cvt_f32_bf16<<<(NB + CT - 1) / CT, CT, 0, stream>>>(key_,  B1, NB);
    cvt_f32_bf16<<<(NB + CT - 1) / CT, CT, 0, stream>>>(value, B2, NB);
    cvt_f32_bf16<<<(NW + CT - 1) / CT, CT, 0, stream>>>(W_Q, WQ, NW);
    cvt_f32_bf16<<<(NW + CT - 1) / CT, CT, 0, stream>>>(W_K, WK, NW);
    cvt_f32_bf16<<<(NW + CT - 1) / CT, CT, 0, stream>>>(W_V, WV, NW);
    cvt_f32_bf16<<<(NW + CT - 1) / CT, CT, 0, stream>>>(W_O, WO, NW);

    dim3 ggrid(M / 64, D_MODEL / 256);
    const float qscale = 0.125f;   // 1/sqrt(DK)
    // Qp = (x@W_Q^T + b_Q) * qscale  -> B3
    gemm_xwT<true><<<ggrid, 256, 0, stream>>>(B0, WQ, b_Q, B3, M, D_MODEL, D_MODEL, qscale);
    // Kp -> B0 (old query bf16 dead)
    gemm_xwT<true><<<ggrid, 256, 0, stream>>>(B1, WK, b_K, B0, M, D_MODEL, D_MODEL, 1.0f);
    // Vp -> B1 (old key bf16 dead)
    gemm_xwT<true><<<ggrid, 256, 0, stream>>>(B2, WV, b_V, B1, M, D_MODEL, D_MODEL, 1.0f);

    // attention: ctx -> B2 (old value bf16 dead)
    dim3 agrid(SEQ / 64, NHEAD, BATCH);
    attn_kernel<<<agrid, 128, 0, stream>>>(B3, B0, B1, B2);

    // output projection -> d_out (fp32)
    gemm_xwT<false><<<ggrid, 256, 0, stream>>>(B2, WO, b_O, (float*)d_out,
                                               M, D_MODEL, D_MODEL, 1.0f);
}
